// FEDformerHeadCustom_53635551593046
// MI455X (gfx1250) — compile-verified, matrix operand order checked by reading, not run on hardware
//
#include <hip/hip_runtime.h>
#include <hip/hip_bf16.h>

// ---------------------------------------------------------------------------
// FEDformer head for MI455X (gfx1250, wave32, WMMA).
// All heavy math (DFT, sparse iDFT, FFNs) is routed through one bf16 WMMA
// GEMM kernel (v_wmma_f32_16x16x32_bf16). Elementwise / reduction glue is
// plain VALU. fp32 in/out, bf16 inside the matrix units, f32 accumulation.
// ---------------------------------------------------------------------------

typedef __bf16 bf16t;
typedef bf16t v16bf __attribute__((ext_vector_type(16)));
typedef float v8f   __attribute__((ext_vector_type(8)));

union FragU { v16bf v; unsigned int u[8]; };

static __device__ __forceinline__ unsigned short f2bf(float x) {
  unsigned int u = __float_as_uint(x);
  unsigned int r = (u + 0x7FFFu + ((u >> 16) & 1u)) >> 16;   // RNE
  return (unsigned short)r;
}

#define BMT 64
#define BNT 128
#define BKT 32

// C[M,N] = A[M,K] x B[K,N]  (+bias) (+gelu | +residual), all row-major fp32,
// batched via blockIdx.z with element strides sA/sB/sC/sRes.
__global__ __launch_bounds__(256) void wmma_gemm_kernel(
    const float* __restrict__ A, const float* __restrict__ Bm,
    float* __restrict__ C, const float* __restrict__ bias,
    const float* __restrict__ res,
    int M, int N, int K,
    long long sA, long long sB, long long sC, long long sRes, int epi) {
  int bz = blockIdx.z;
  A  += (size_t)bz * sA;
  Bm += (size_t)bz * sB;
  C  += (size_t)bz * sC;
  const float* resp = res ? res + (size_t)bz * sRes : nullptr;

  int blockM = blockIdx.y * BMT;
  int blockN = blockIdx.x * BNT;

  __shared__ unsigned short As[BMT][BKT + 2];   // [row][k] bf16 bits
  __shared__ unsigned short Bs[BNT][BKT + 2];   // [col][k] bf16 bits (B^T)

  int tid  = threadIdx.x;
  int lane = tid & 31, wave = tid >> 5;
  int wm = wave >> 2, wn = wave & 3;            // 2 x 4 wave grid
  int l16 = lane & 15, hi = lane >> 4;

  v8f acc[2][2];
#pragma unroll
  for (int a = 0; a < 2; ++a)
#pragma unroll
    for (int c2 = 0; c2 < 2; ++c2)
#pragma unroll
      for (int e = 0; e < 8; ++e) acc[a][c2][e] = 0.f;

  int ksteps = (K + BKT - 1) / BKT;
  for (int ks = 0; ks < ksteps; ++ks) {
    int k0 = ks * BKT;
    {   // A tile: 64x32, 8 floats per thread
      int r = tid >> 2, c0 = (tid & 3) * 8;
      int gr = blockM + r;
#pragma unroll
      for (int u = 0; u < 8; ++u) {
        int gc = k0 + c0 + u;
        float v = (gr < M && gc < K) ? A[(size_t)gr * K + gc] : 0.f;
        As[r][c0 + u] = f2bf(v);
      }
    }
    {   // B tile: 32x128, transposed into LDS, 16 floats per thread
      int kk = tid >> 3, n0 = (tid & 7) * 16;
      int gk = k0 + kk;
#pragma unroll
      for (int u = 0; u < 16; ++u) {
        int gn = blockN + n0 + u;
        float v = (gk < K && gn < N) ? Bm[(size_t)gk * N + gn] : 0.f;
        Bs[n0 + u][kk] = f2bf(v);
      }
    }
    __syncthreads();

    FragU fa[2], fb[2];
#pragma unroll
    for (int a = 0; a < 2; ++a) {
      int row = wm * 32 + a * 16 + l16;
#pragma unroll
      for (int p = 0; p < 8; ++p) {
        int kb = (p < 4) ? (hi * 8 + 2 * p) : (16 + hi * 8 + 2 * (p - 4));
        fa[a].u[p] = *(const unsigned int*)&As[row][kb];
      }
    }
#pragma unroll
    for (int c2 = 0; c2 < 2; ++c2) {
      int col = wn * 32 + c2 * 16 + l16;
#pragma unroll
      for (int p = 0; p < 8; ++p) {
        int kb = (p < 4) ? (hi * 8 + 2 * p) : (16 + hi * 8 + 2 * (p - 4));
        fb[c2].u[p] = *(const unsigned int*)&Bs[col][kb];
      }
    }
#pragma unroll
    for (int a = 0; a < 2; ++a)
#pragma unroll
      for (int c2 = 0; c2 < 2; ++c2)
        acc[a][c2] = __builtin_amdgcn_wmma_f32_16x16x32_bf16(
            false, fa[a].v, false, fb[c2].v, (short)0, acc[a][c2], false, false);
    __syncthreads();
  }

#pragma unroll
  for (int a = 0; a < 2; ++a)
#pragma unroll
    for (int c2 = 0; c2 < 2; ++c2)
#pragma unroll
      for (int e = 0; e < 8; ++e) {
        int gr = blockM + wm * 32 + a * 16 + e + 8 * hi;
        int gn = blockN + wn * 32 + c2 * 16 + l16;
        if (gr < M && gn < N) {
          float v = acc[a][c2][e];
          if (bias) v += bias[gn];
          if (epi == 1) v = 0.5f * v * (1.f + erff(v * 0.70710678f));       // exact GELU
          else if (epi == 2) v += resp[(size_t)gr * N + gn];                // residual
          C[(size_t)gr * N + gn] = v;
        }
      }
}

// ------------------------------ glue kernels -------------------------------

#define NB(n) (((n) + 255) / 256)

// series decomposition: moving average k=25, zero padded
__global__ void decomp_kernel(const float* __restrict__ x, float* __restrict__ s,
                              float* __restrict__ t_, int total) {
  int id = blockIdx.x * blockDim.x + threadIdx.x;
  if (id >= total) return;
  int c = id & 7;
  int t = (id >> 3) & 1023;
  int b = id >> 13;
  float acc = 0.f;
  int j0 = t - 12, j1 = t + 12;
#pragma unroll 1
  for (int j = j0; j <= j1; ++j)
    if (j >= 0 && j < 1024) acc += x[(((size_t)b << 10) + j) * 8 + c];
  float tm = acc * (1.f / 25.f);
  t_[id] = tm;
  s_ignore:
  s[id] = x[id] - tm;
}

// sez = s @ W(8x512) + b
__global__ void embed_kernel(const float* __restrict__ s, const float* __restrict__ W,
                             const float* __restrict__ bias, float* __restrict__ out,
                             int rows) {
  int id = blockIdx.x * blockDim.x + threadIdx.x;
  if (id >= rows * 512) return;
  int d = id & 511, row = id >> 9;
  float acc = bias[d];
#pragma unroll
  for (int c = 0; c < 8; ++c) acc += s[(size_t)row * 8 + c] * W[c * 512 + d];
  out[id] = acc;
}

__global__ void rmsnorm_kernel(const float* __restrict__ in, float* __restrict__ out,
                               const float* __restrict__ w, int rows) {
  int row = blockIdx.x;
  if (row >= rows) return;
  const float* x = in + (size_t)row * 512;
  __shared__ float red[256];
  float ss = 0.f;
  for (int i = threadIdx.x; i < 512; i += 256) { float v = x[i]; ss += v * v; }
  red[threadIdx.x] = ss; __syncthreads();
  for (int s = 128; s > 0; s >>= 1) {
    if (threadIdx.x < s) red[threadIdx.x] += red[threadIdx.x + s];
    __syncthreads();
  }
  float sc = rsqrtf(red[0] * (1.f / 512.f) + 1e-5f);
  for (int i = threadIdx.x; i < 512; i += 256)
    out[(size_t)row * 512 + i] = x[i] * sc * w[i];
}

// base += rmsnorm(y) * w
__global__ void rmsnorm_add_kernel(float* __restrict__ base, const float* __restrict__ y,
                                   const float* __restrict__ w, int rows) {
  int row = blockIdx.x;
  if (row >= rows) return;
  const float* x = y + (size_t)row * 512;
  __shared__ float red[256];
  float ss = 0.f;
  for (int i = threadIdx.x; i < 512; i += 256) { float v = x[i]; ss += v * v; }
  red[threadIdx.x] = ss; __syncthreads();
  for (int s = 128; s > 0; s >>= 1) {
    if (threadIdx.x < s) red[threadIdx.x] += red[threadIdx.x + s];
    __syncthreads();
  }
  float sc = rsqrtf(red[0] * (1.f / 512.f) + 1e-5f);
  for (int i = threadIdx.x; i < 512; i += 256)
    base[(size_t)row * 512 + i] += x[i] * sc * w[i];
}

// xd = tgt + rmsnorm(y)*nw * gate[b]
__global__ void combine_kernel(const float* __restrict__ tgt, const float* __restrict__ y,
                               const float* __restrict__ nw, const float* __restrict__ gate,
                               float* __restrict__ xd, int rows) {
  int row = blockIdx.x;
  if (row >= rows) return;
  int b = row / 336;
  const float* x = y + (size_t)row * 512;
  __shared__ float red[256];
  float ss = 0.f;
  for (int i = threadIdx.x; i < 512; i += 256) { float v = x[i]; ss += v * v; }
  red[threadIdx.x] = ss; __syncthreads();
  for (int s = 128; s > 0; s >>= 1) {
    if (threadIdx.x < s) red[threadIdx.x] += red[threadIdx.x + s];
    __syncthreads();
  }
  float sc = rsqrtf(red[0] * (1.f / 512.f) + 1e-5f);
  for (int i = threadIdx.x; i < 512; i += 256)
    xd[(size_t)row * 512 + i] =
        tgt[(size_t)row * 512 + i] + x[i] * sc * nw[i] * gate[(size_t)b * 512 + i];
}

// DFT basis rows: row 2f = cos(2*pi*f*t/T), row 2f+1 = -sin(...)   (rfft)
__global__ void dftbasis_kernel(float* __restrict__ dftb, int T, int F) {
  int id = blockIdx.x * blockDim.x + threadIdx.x;
  if (id >= F * T) return;
  int f = id / T, t = id % T;
  long long ph = ((long long)f * t) % T;
  float th = 6.283185307179586f * (float)ph / (float)T;
  dftb[(size_t)(2 * f) * T + t]     =  cosf(th);
  dftb[(size_t)(2 * f + 1) * T + t] = -sinf(th);
}

// mag[b,j] = sum_c Re^2+Im^2 at frequency f=j+1 ; Xf layout [b][2F][D]
__global__ void mag_kernel(const float* __restrict__ Xf, float* __restrict__ mag,
                           int F2, int nf) {
  int id = blockIdx.x * blockDim.x + threadIdx.x;
  if (id >= 32 * nf) return;
  int b = id / nf, j = id % nf, f = j + 1;
  const float* re = Xf + ((size_t)b * F2 + 2 * f) * 512;
  const float* im = re + 512;
  float acc = 0.f;
  for (int c = 0; c < 512; ++c) acc += re[c] * re[c] + im[c] * im[c];
  mag[(size_t)b * nf + j] = acc;
}

// top-64 (descending) per batch; stores frequency index (j+1)
__global__ void topk_kernel(const float* __restrict__ mag, int* __restrict__ idxo, int nf) {
  int b = blockIdx.x;
  __shared__ float vals[512];
  __shared__ float rv[256];
  __shared__ int   ri[256];
  for (int i = threadIdx.x; i < 512; i += 256)
    vals[i] = (i < nf) ? mag[(size_t)b * nf + i] : -3.4e38f;
  __syncthreads();
  for (int it = 0; it < 64; ++it) {
    float bv = -3.4e38f; int bi = 0;
    for (int i = threadIdx.x; i < 512; i += 256)
      if (vals[i] > bv) { bv = vals[i]; bi = i; }
    rv[threadIdx.x] = bv; ri[threadIdx.x] = bi; __syncthreads();
    for (int s = 128; s > 0; s >>= 1) {
      if (threadIdx.x < s && rv[threadIdx.x + s] > rv[threadIdx.x]) {
        rv[threadIdx.x] = rv[threadIdx.x + s]; ri[threadIdx.x] = ri[threadIdx.x + s];
      }
      __syncthreads();
    }
    if (threadIdx.x == 0) { idxo[b * 64 + it] = ri[0] + 1; vals[ri[0]] = -3.4e38f; }
    __syncthreads();
  }
}

// Yfat[b][2k][c], [2k+1][c] = complex (Xf[b,f] * Wc[k])
__global__ void gather_kernel(const float* __restrict__ Xf, const int* __restrict__ idx,
                              const float* __restrict__ w, float* __restrict__ Yfat, int F2) {
  int id = blockIdx.x * blockDim.x + threadIdx.x;
  if (id >= 32 * 64 * 512) return;
  int c = id & 511, k = (id >> 9) & 63, b = id >> 15;
  int f = idx[b * 64 + k];
  const float* re = Xf + ((size_t)b * F2 + 2 * f) * 512;
  const float* im = re + 512;
  float wr = w[((size_t)k * 512 + c) * 2 + 0];
  float wi = w[((size_t)k * 512 + c) * 2 + 1];
  float xr = re[c], xi = im[c];
  Yfat[((size_t)b * 128 + 2 * k) * 512 + c]     = xr * wr - xi * wi;
  Yfat[((size_t)b * 128 + 2 * k + 1) * 512 + c] = xr * wi + xi * wr;
}

// iDFT basis: ybas[b][t][2k,2k+1] = scale*cos, -scale*sin ; scale handles Nyquist
__global__ void ybasis_kernel(const int* __restrict__ idx, float* __restrict__ ybas, int T) {
  int id = blockIdx.x * blockDim.x + threadIdx.x;
  if (id >= 32 * T * 64) return;
  int k = id & 63, t = (id >> 6) % T, b = id / (64 * T);
  int f = idx[b * 64 + k];
  long long ph = ((long long)f * t) % T;
  float th = 6.283185307179586f * (float)ph / (float)T;
  float sc = (2 * f == T) ? (1.0f / T) : (2.0f / T);
  float* row = ybas + ((size_t)b * T + t) * 128;
  row[2 * k]     =  sc * cosf(th);
  row[2 * k + 1] = -sc * sinf(th);
}

// m[b,d] = mean_t sez[b,t,d]
__global__ void mean_kernel(const float* __restrict__ sez, float* __restrict__ m) {
  int id = blockIdx.x * blockDim.x + threadIdx.x;
  if (id >= 32 * 512) return;
  int b = id >> 9, d = id & 511;
  float acc = 0.f;
  for (int t = 0; t < 1024; ++t) acc += sez[(((size_t)b << 10) + t) * 512 + d];
  m[id] = acc * (1.f / 1024.f);
}

// gate = sigmoid(rmsnorm(m)*nx @ W + b)
__global__ void cross_kernel(const float* __restrict__ m, const float* __restrict__ nx,
                             const float* __restrict__ W, const float* __restrict__ bias,
                             float* __restrict__ gate) {
  int b = blockIdx.x;
  __shared__ float mn[512];
  __shared__ float red[256];
  float ss = 0.f;
  for (int i = threadIdx.x; i < 512; i += 256) { float v = m[(size_t)b * 512 + i]; ss += v * v; }
  red[threadIdx.x] = ss; __syncthreads();
  for (int s = 128; s > 0; s >>= 1) {
    if (threadIdx.x < s) red[threadIdx.x] += red[threadIdx.x + s];
    __syncthreads();
  }
  float sc = rsqrtf(red[0] * (1.f / 512.f) + 1e-5f);
  for (int i = threadIdx.x; i < 512; i += 256)
    mn[i] = m[(size_t)b * 512 + i] * sc * nx[i];
  __syncthreads();
  for (int d = threadIdx.x; d < 512; d += 256) {
    float acc = bias[d];
    for (int k = 0; k < 512; ++k) acc += mn[k] * W[(size_t)k * 512 + d];
    gate[(size_t)b * 512 + d] = 1.f / (1.f + __expf(-acc));
  }
}

__global__ void tgt_kernel(const float* __restrict__ qpos, const float* __restrict__ sez,
                           float* __restrict__ tgt) {
  int id = blockIdx.x * blockDim.x + threadIdx.x;
  if (id >= 32 * 336 * 512) return;
  int d = id & 511, p = (id >> 9) % 336, b = id / (336 * 512);
  tgt[id] = qpos[(size_t)p * 512 + d] + sez[(((size_t)b << 10) + 1023) * 512 + d];
}

// seasonal projection D->8
__global__ void seas_kernel(const float* __restrict__ xd, const float* __restrict__ W,
                            const float* __restrict__ bias, float* __restrict__ out, int rows) {
  int id = blockIdx.x * blockDim.x + threadIdx.x;
  if (id >= rows * 8) return;
  int co = id & 7, row = id >> 3;
  const float* x = xd + (size_t)row * 512;
  float acc = bias[co];
  for (int d = 0; d < 512; ++d) acc += x[d] * W[d * 8 + co];
  out[id] = acc;
}

// trend path: last timestep only
__global__ void trlast_kernel(const float* __restrict__ t_, const float* __restrict__ W,
                              const float* __restrict__ bias, float* __restrict__ out) {
  int id = blockIdx.x * blockDim.x + threadIdx.x;
  if (id >= 32 * 512) return;
  int b = id >> 9, d = id & 511;
  float acc = bias[d];
#pragma unroll
  for (int c = 0; c < 8; ++c)
    acc += t_[(((size_t)b << 10) + 1023) * 8 + c] * W[c * 512 + d];
  out[id] = acc;
}

__global__ void trhid_kernel(const float* __restrict__ trl, const float* __restrict__ W,
                             const float* __restrict__ bias, float* __restrict__ out) {
  int id = blockIdx.x * blockDim.x + threadIdx.x;
  if (id >= 32 * 2048) return;
  int b = id >> 11, h = id & 2047;
  float acc = bias[h];
  for (int d = 0; d < 512; ++d) acc += trl[(size_t)b * 512 + d] * W[(size_t)d * 2048 + h];
  out[id] = 0.5f * acc * (1.f + erff(acc * 0.70710678f));
}

__global__ void trout_kernel(const float* __restrict__ trh, const float* __restrict__ W,
                             const float* __restrict__ bias, float* __restrict__ out) {
  int id = blockIdx.x * blockDim.x + threadIdx.x;
  if (id >= 32 * 8) return;
  int b = id >> 3, co = id & 7;
  float acc = bias[co];
  for (int h = 0; h < 2048; ++h) acc += trh[(size_t)b * 2048 + h] * W[h * 8 + co];
  out[id] = acc;
}

__global__ void mix_kernel(const float* __restrict__ seas, const float* __restrict__ trout,
                           const float* __restrict__ mixw, const float* __restrict__ mixb,
                           float* __restrict__ out, int rows) {
  int id = blockIdx.x * blockDim.x + threadIdx.x;
  if (id >= rows * 8) return;
  int co = id & 7, row = id >> 3, b = row / 336;
  float acc = mixb[co];
#pragma unroll
  for (int c = 0; c < 8; ++c) {
    float v = seas[(size_t)row * 8 + c] + trout[b * 8 + c];
    acc += v * mixw[c * 8 + co];
  }
  out[id] = acc;
}

// ---------------------------------------------------------------------------

extern "C" void kernel_launch(void* const* d_in, const int* in_sizes, int n_in,
                              void* d_out, int out_size, void* d_ws, size_t ws_size,
                              hipStream_t stream) {
  const int B = 32, L = 1024, D = 512, FF = 2048, PRED = 336;
  const int T_ENC = 1024, F_ENC = 513, T_DEC = 336, F_DEC = 169;
  const int F2E = 2 * F_ENC, F2D = 2 * F_DEC;
  const int CH = 4096;                         // FFN row chunk

  const float* x          = (const float*)d_in[0];
  const float* enc_in_w   = (const float*)d_in[1];
  const float* enc_in_b   = (const float*)d_in[2];
  const float* dec_in_w   = (const float*)d_in[3];
  const float* dec_in_b   = (const float*)d_in[4];
  const float* enc_freq_w = (const float*)d_in[5];
  const float* enc_freq_nw= (const float*)d_in[6];
  const float* enc_n1     = (const float*)d_in[7];
  const float* enc_n2     = (const float*)d_in[8];
  const float* enc_w1     = (const float*)d_in[9];
  const float* enc_b1     = (const float*)d_in[10];
  const float* enc_w2     = (const float*)d_in[11];
  const float* enc_b2     = (const float*)d_in[12];
  const float* dec_freq_w = (const float*)d_in[13];
  const float* dec_freq_nw= (const float*)d_in[14];
  const float* dec_n1     = (const float*)d_in[15];
  const float* dec_n2     = (const float*)d_in[16];
  const float* dec_nx     = (const float*)d_in[17];
  const float* dec_cross_w= (const float*)d_in[18];
  const float* dec_cross_b= (const float*)d_in[19];
  const float* dec_w1     = (const float*)d_in[20];
  const float* dec_b1     = (const float*)d_in[21];
  const float* dec_w2     = (const float*)d_in[22];
  const float* dec_b2     = (const float*)d_in[23];
  const float* seas_w     = (const float*)d_in[24];
  const float* seas_b     = (const float*)d_in[25];
  const float* tr_w1      = (const float*)d_in[26];
  const float* tr_b1      = (const float*)d_in[27];
  const float* tr_w2      = (const float*)d_in[28];
  const float* tr_b2      = (const float*)d_in[29];
  const float* mix_w      = (const float*)d_in[30];
  const float* mix_b      = (const float*)d_in[31];
  const float* qpos       = (const float*)d_in[32];

  // ---- workspace layout (floats) ----
  float* w = (float*)d_ws;
  size_t o = 0;
  auto alloc = [&](size_t n) { float* p = w + o; o += n; return p; };
  float* sez    = alloc((size_t)B * L * D);          // 16.8M
  float* xn     = alloc((size_t)B * L * D);          // normed in / iDFT out
  float* Xf     = alloc((size_t)B * F2E * D);        // [b][2F][D]
  float* hidden = alloc((size_t)CH * FF);            // FFN chunk hidden
  float* dftb   = alloc((size_t)F2E * T_ENC);        // DFT basis (regen per use)
  float* ybas   = alloc((size_t)B * T_ENC * 128);    // iDFT basis
  float* Yfat   = alloc((size_t)B * 128 * D);        // selected modes (Re/Im rows)
  float* sbuf   = alloc((size_t)B * L * 8);
  float* tbuf   = alloc((size_t)B * L * 8);
  float* mag    = alloc((size_t)B * 512);
  int*   idx    = (int*)alloc((size_t)B * 64);
  float* mbuf   = alloc((size_t)B * D);
  float* gate   = alloc((size_t)B * D);
  float* tgt    = alloc((size_t)B * PRED * D);
  float* xd     = alloc((size_t)B * PRED * D);
  float* seas   = alloc((size_t)B * PRED * 8);
  float* trl    = alloc((size_t)B * D);
  float* trh    = alloc((size_t)B * FF);
  float* tro    = alloc((size_t)B * 8);
  (void)ws_size; (void)n_in; (void)in_sizes; (void)out_size;

  auto gemm = [&](const float* A, const float* Bm, float* C, const float* bias,
                  const float* res, int M, int N, int K, long long sA, long long sB,
                  long long sC, long long sRes, int batch, int epi) {
    dim3 g((N + BNT - 1) / BNT, (M + BMT - 1) / BMT, batch);
    wmma_gemm_kernel<<<g, 256, 0, stream>>>(A, Bm, C, bias, res, M, N, K,
                                            sA, sB, sC, sRes, epi);
  };

  // ---- series decomposition + encoder embed ----
  int totBLC = B * L * 8;
  decomp_kernel<<<NB(totBLC), 256, 0, stream>>>(x, sbuf, tbuf, totBLC);
  embed_kernel<<<NB(B * L * D), 256, 0, stream>>>(sbuf, enc_in_w, enc_in_b, sez, B * L);

  const int rowsE = B * L;

  // ---- encoder layers ----
  for (int i = 0; i < 2; ++i) {
    const float* n1 = enc_n1 + i * D;
    const float* n2 = enc_n2 + i * D;
    const float* fw = enc_freq_w + (size_t)i * 64 * D * 2;
    const float* fnw = enc_freq_nw + i * D;
    const float* w1 = enc_w1 + (size_t)i * D * FF;
    const float* b1 = enc_b1 + (size_t)i * FF;
    const float* w2 = enc_w2 + (size_t)i * FF * D;
    const float* b2 = enc_b2 + (size_t)i * D;

    // Fourier block
    rmsnorm_kernel<<<rowsE, 256, 0, stream>>>(sez, xn, n1, rowsE);
    dftbasis_kernel<<<NB(F_ENC * T_ENC), 256, 0, stream>>>(dftb, T_ENC, F_ENC);
    gemm(dftb, xn, Xf, nullptr, nullptr, F2E, D, T_ENC,
         0, (long long)L * D, (long long)F2E * D, 0, B, 0);
    mag_kernel<<<NB(B * (F_ENC - 1)), 256, 0, stream>>>(Xf, mag, F2E, F_ENC - 1);
    topk_kernel<<<B, 256, 0, stream>>>(mag, idx, F_ENC - 1);
    gather_kernel<<<NB(B * 64 * D), 256, 0, stream>>>(Xf, idx, fw, Yfat, F2E);
    ybasis_kernel<<<NB(B * T_ENC * 64), 256, 0, stream>>>(idx, ybas, T_ENC);
    gemm(ybas, Yfat, xn, nullptr, nullptr, T_ENC, D, 128,
         (long long)T_ENC * 128, (long long)128 * D, (long long)L * D, 0, B, 0);
    rmsnorm_add_kernel<<<rowsE, 256, 0, stream>>>(sez, xn, fnw, rowsE);

    // FFN (chunked)
    rmsnorm_kernel<<<rowsE, 256, 0, stream>>>(sez, xn, n2, rowsE);
    for (int r0 = 0; r0 < rowsE; r0 += CH) {
      int rc = (rowsE - r0 < CH) ? (rowsE - r0) : CH;
      gemm(xn + (size_t)r0 * D, w1, hidden, b1, nullptr, rc, FF, D, 0, 0, 0, 0, 1, 1);
      gemm(hidden, w2, sez + (size_t)r0 * D, b2, sez + (size_t)r0 * D,
           rc, D, FF, 0, 0, 0, 0, 1, 2);
    }
  }

  // ---- decoder prep ----
  mean_kernel<<<NB(B * D), 256, 0, stream>>>(sez, mbuf);
  cross_kernel<<<B, 256, 0, stream>>>(mbuf, dec_nx, dec_cross_w, dec_cross_b, gate);
  tgt_kernel<<<NB(B * PRED * D), 256, 0, stream>>>(qpos, sez, tgt);

  const int rowsD = B * PRED;

  // ---- decoder Fourier block (T=336) ----
  rmsnorm_kernel<<<rowsD, 256, 0, stream>>>(tgt, xn, dec_n1, rowsD);
  dftbasis_kernel<<<NB(F_DEC * T_DEC), 256, 0, stream>>>(dftb, T_DEC, F_DEC);
  gemm(dftb, xn, Xf, nullptr, nullptr, F2D, D, T_DEC,
       0, (long long)T_DEC * D, (long long)F2D * D, 0, B, 0);
  mag_kernel<<<NB(B * (F_DEC - 1)), 256, 0, stream>>>(Xf, mag, F2D, F_DEC - 1);
  topk_kernel<<<B, 256, 0, stream>>>(mag, idx, F_DEC - 1);
  gather_kernel<<<NB(B * 64 * D), 256, 0, stream>>>(Xf, idx, dec_freq_w, Yfat, F2D);
  ybasis_kernel<<<NB(B * T_DEC * 64), 256, 0, stream>>>(idx, ybas, T_DEC);
  gemm(ybas, Yfat, xn, nullptr, nullptr, T_DEC, D, 128,
       (long long)T_DEC * 128, (long long)128 * D, (long long)T_DEC * D, 0, B, 0);
  combine_kernel<<<rowsD, 256, 0, stream>>>(tgt, xn, dec_freq_nw, gate, xd, rowsD);

  // ---- decoder FFN ----
  rmsnorm_kernel<<<rowsD, 256, 0, stream>>>(xd, xn, dec_n2, rowsD);
  for (int r0 = 0; r0 < rowsD; r0 += CH) {
    int rc = (rowsD - r0 < CH) ? (rowsD - r0) : CH;
    gemm(xn + (size_t)r0 * D, dec_w1, hidden, dec_b1, nullptr, rc, FF, D, 0, 0, 0, 0, 1, 1);
    gemm(hidden, dec_w2, xd + (size_t)r0 * D, dec_b2, xd + (size_t)r0 * D,
         rc, D, FF, 0, 0, 0, 0, 1, 2);
  }

  // ---- output heads ----
  seas_kernel<<<NB(rowsD * 8), 256, 0, stream>>>(xd, seas_w, seas_b, seas, rowsD);
  trlast_kernel<<<NB(B * D), 256, 0, stream>>>(tbuf, dec_in_w, dec_in_b, trl);
  trhid_kernel<<<NB(B * FF), 256, 0, stream>>>(trl, tr_w1, tr_b1, trh);
  trout_kernel<<<NB(B * 8), 256, 0, stream>>>(trh, tr_w2, tr_b2, tro);
  mix_kernel<<<NB(rowsD * 8), 256, 0, stream>>>(seas, tro, mix_w, mix_b,
                                                (float*)d_out, rowsD);
}
